// ConvFunc_29437705846800
// MI455X (gfx1250) — matrix operand order repaired and checked by reading, not compiled, not run on hardware
//
#include <hip/hip_runtime.h>
#include <cstdint>

typedef __attribute__((ext_vector_type(16))) _Float16 v16h;
typedef __attribute__((ext_vector_type(8)))  _Float16 v8h;
typedef __attribute__((ext_vector_type(8)))  float    v8f;

#define DIM 128
#define LDP 40    // padded LDS row for 32-wide K chunk (f16): 80B stride
#define LDR 136   // padded LDS row for 128-wide channel row (f16): 272B stride

__device__ __forceinline__ v16h cat16(v8h lo, v8h hi) {
  return __builtin_shufflevector(lo, hi, 0,1,2,3,4,5,6,7,8,9,10,11,12,13,14,15);
}

// async 16B global->LDS copy (ASYNCcnt-tracked, ISA §10/§15.18)
__device__ __forceinline__ void async_copy16(unsigned lds_off, const void* gaddr) {
  asm volatile("global_load_async_to_lds_b128 %0, %1, off"
               :: "v"(lds_off), "v"(gaddr) : "memory");
}
__device__ __forceinline__ void wait_async0() {
  asm volatile("s_wait_asynccnt 0x0" ::: "memory");
}

// -------- Pass 0a: repack per-sample conv weights q -> f16 wq[b][tap][o][i]
__global__ __launch_bounds__(256) void wrepack_kernel(
    const float* __restrict__ q, _Float16* __restrict__ wq)
{
  size_t e = (size_t)blockIdx.x * 256 + threadIdx.x;   // < B*9*128*128
  int i  = (int)(e & 127);
  int o  = (int)((e >> 7) & 127);
  int bp = (int)(e >> 14);                             // b*9 + p
  int p  = bp % 9;
  int b  = bp / 9;
  wq[e] = (_Float16)q[(size_t)b * (DIM*DIM*9 + DIM) + o*(DIM*9) + i*9 + p];
}

// -------- Pass 0b: proj_w -> f16 (64KB, stays L2-hot; enables async A staging)
__global__ __launch_bounds__(256) void pwrepack_kernel(
    const float* __restrict__ pw, _Float16* __restrict__ pwh)
{
  int e = blockIdx.x * 256 + threadIdx.x;              // < 128*256
  pwh[e] = (_Float16)pw[e];
}

// -------- Pass 1: inp[b][hw][o] = f16( proj_w(128x256) x cat[b](256x1024) + proj_b )
__global__ __launch_bounds__(256) void proj_kernel(
    const float* __restrict__ lhs, const float* __restrict__ rhs,
    const _Float16* __restrict__ pwh, const float* __restrict__ pb,
    _Float16* __restrict__ inp)
{
  const int b    = blockIdx.x >> 3;
  const int hw0  = (blockIdx.x & 7) << 7;       // 128-wide spatial tile
  const int tid  = threadIdx.x;
  const int wave = tid >> 5, lane = tid & 31;
  const int half = lane >> 4, lr = lane & 15;

  __shared__ _Float16 As[2][DIM * LDP];         // proj_w chunk [o][kc]
  __shared__ _Float16 Bs[2][DIM * LDP];         // cat chunk    [hw][kc]

  v8f    acc[8] = {};
  float4 rb4[4];

  // A chunk: async DMA straight from f16 proj_w (no VGPR round trip)
  auto stage_a = [&](int k0, int buf) {
    #pragma unroll
    for (int j = 0; j < 2; ++j) {
      int i = tid + j * 256;                    // 0..511 16B chunks
      int m = i >> 2, c8 = i & 3;
      async_copy16((unsigned)(uintptr_t)&As[buf][m * LDP + c8 * 8],
                   pwh + m * 256 + k0 + c8 * 8);
    }
  };
  // B chunk: prefetch next chunk early so the real loads hit L2/WGP$
  auto prefetchB = [&](int k0) {
    const float* src = (k0 < DIM) ? lhs : rhs;
    const int    cb  = k0 & (DIM - 1);
    #pragma unroll
    for (int j = 0; j < 4; ++j) {
      int i = tid + j * 256;
      int kc = i >> 5, hw4 = i & 31;
      __builtin_prefetch(src + ((b * DIM + cb + kc) << 10) + hw0 + hw4 * 4, 0, 1);
    }
  };
  auto loadB = [&](int k0) {
    const float* src = (k0 < DIM) ? lhs : rhs;  // uniform per chunk
    const int    cb  = k0 & (DIM - 1);
    #pragma unroll
    for (int j = 0; j < 4; ++j) {
      int i = tid + j * 256;                    // 0..1023 float4 chunks
      int kc = i >> 5, hw4 = i & 31;
      rb4[j] = *(const float4*)(src + ((b * DIM + cb + kc) << 10) + hw0 + hw4 * 4);
    }
  };
  auto storeB = [&](int buf) {
    #pragma unroll
    for (int j = 0; j < 4; ++j) {
      int i = tid + j * 256;
      int kc = i >> 5, hw4 = i & 31;
      Bs[buf][(hw4 * 4 + 0) * LDP + kc] = (_Float16)rb4[j].x;  // transpose scatter
      Bs[buf][(hw4 * 4 + 1) * LDP + kc] = (_Float16)rb4[j].y;
      Bs[buf][(hw4 * 4 + 2) * LDP + kc] = (_Float16)rb4[j].z;
      Bs[buf][(hw4 * 4 + 3) * LDP + kc] = (_Float16)rb4[j].w;
    }
  };

  stage_a(0, 0);
  loadB(0);
  storeB(0);
  wait_async0();
  __syncthreads();

  for (int ks = 0; ks < 8; ++ks) {
    const int buf = ks & 1;
    if (ks < 7) {
      stage_a((ks + 1) * 32, buf ^ 1);          // DMA next A under the WMMAs
      prefetchB((ks + 1) * 32);                 // warm next B into L2/WGP$
    }

    const _Float16* arow = &As[buf][(wave * 16 + lr) * LDP];
    v16h a = cat16(*(const v8h*)(arow + 16 * half),
                   *(const v8h*)(arow + 16 * half + 8));
    v16h bmv[8];
    #pragma unroll
    for (int nt = 0; nt < 8; ++nt) {
      const _Float16* brow = &Bs[buf][(nt * 16 + lr) * LDP + 16 * half];
      bmv[nt] = cat16(*(const v8h*)brow, *(const v8h*)(brow + 8));
    }
    #pragma unroll
    for (int nt = 0; nt < 8; ++nt)
      acc[nt] = __builtin_amdgcn_wmma_f32_16x16x32_f16(
          false, a, false, bmv[nt], (short)0, acc[nt], false, false);

    if (ks < 7) {
      loadB((ks + 1) * 32);                     // short-latency (prefetched) loads
      storeB(buf ^ 1);
      wait_async0();                            // A DMA done
      __syncthreads();
    }
  }

  float bias[8];
  #pragma unroll
  for (int r = 0; r < 8; ++r) bias[r] = pb[wave * 16 + 8 * half + r];

  #pragma unroll
  for (int nt = 0; nt < 8; ++nt) {
    int hw = hw0 + nt * 16 + lr;
    v8h ph;
    #pragma unroll
    for (int r = 0; r < 8; ++r) ph[r] = (_Float16)(acc[nt][r] + bias[r]);
    *(v8h*)(inp + ((size_t)((b << 10) + hw)) * DIM + wave * 16 + 8 * half) = ph;
  }
}

// -------- Pass 2: out[b][o][h][w] = sum_{tap,i} W[o][i][tap]*inp[h+dh][w+dw][i] + bias
__global__ __launch_bounds__(256) void conv_kernel(
    const float* __restrict__ q, const _Float16* __restrict__ wq,
    const _Float16* __restrict__ inp, float* __restrict__ out)
{
  const int b    = blockIdx.x >> 3;
  const int h0   = (blockIdx.x & 7) << 2;       // 4 output rows per block
  const int tid  = threadIdx.x;
  const int wave = tid >> 5, lane = tid & 31;
  const int half = lane >> 4, lr = lane & 15;
  const int wdim = DIM * DIM * 9;

  __shared__ _Float16 iS[192 * LDR];            // 6 halo rows x 32 cols, all 128 ch
  __shared__ _Float16 wS[2][DIM * LDR];         // one tap's weights: [o][i]

  v8f acc[8] = {};

  auto stage_w = [&](int p, int buf) {
    const _Float16* wqb = wq + (((size_t)b * 9 + p) << 14);
    #pragma unroll
    for (int j = 0; j < 8; ++j) {
      int i = tid + j * 256;                    // coalesced 16B per lane
      int o = i >> 4, c8 = i & 15;
      async_copy16((unsigned)(uintptr_t)&wS[buf][o * LDR + c8 * 8], wqb + i * 8);
    }
  };

  // halo window: rows h0-1..h0+4; async-copy in-image rows, zero-store the rest
  for (int i = tid; i < 192 * 16; i += 256) {
    int sp = i >> 4, c8 = i & 15;
    int hs = h0 - 1 + (sp >> 5);
    int ws = sp & 31;
    if (hs >= 0 && hs < 32) {
      async_copy16((unsigned)(uintptr_t)&iS[sp * LDR + c8 * 8],
                   inp + ((size_t)((b << 10) + (hs << 5) + ws)) * DIM + c8 * 8);
    } else {
      v8h z = {};
      *(v8h*)(&iS[sp * LDR + c8 * 8]) = z;
    }
  }
  stage_w(0, 0);
  wait_async0();
  __syncthreads();

  for (int p = 0; p < 9; ++p) {
    if (p < 8) stage_w(p + 1, (p + 1) & 1);     // DMA next tap under the WMMAs
    const int buf = p & 1;
    const int dh = p / 3 - 1, dw = p % 3 - 1;

    const _Float16* arow = &wS[buf][(wave * 16 + lr) * LDR];
    v16h av[4];
    #pragma unroll
    for (int kq = 0; kq < 4; ++kq)
      av[kq] = cat16(*(const v8h*)(arow + kq * 32 + 16 * half),
                     *(const v8h*)(arow + kq * 32 + 16 * half + 8));

    #pragma unroll
    for (int nt = 0; nt < 8; ++nt) {
      int n  = nt * 16 + lr;                    // position within 4-row window
      int ww = (n & 31) + dw;
      bool ok = (ww >= 0) && (ww < 32);
      int spb = ((n >> 5) + dh + 1) * 32 + (ok ? ww : 0);
      const _Float16* brow = &iS[spb * LDR + 16 * half];
      v16h bm[4] = {};
      if (ok) {                                 // edge lanes keep zero fragments
        #pragma unroll
        for (int kq = 0; kq < 4; ++kq)
          bm[kq] = cat16(*(const v8h*)(brow + kq * 32),
                         *(const v8h*)(brow + kq * 32 + 8));
      }
      #pragma unroll
      for (int kq = 0; kq < 4; ++kq)
        acc[nt] = __builtin_amdgcn_wmma_f32_16x16x32_f16(
            false, av[kq], false, bm[kq], (short)0, acc[nt], false, false);
    }

    if (p < 8) {
      wait_async0();                            // next tap's DMA done
      __syncthreads();
    }
  }

  const float* qb = q + (size_t)b * (wdim + DIM);
  float bias[8];
  #pragma unroll
  for (int r = 0; r < 8; ++r) bias[r] = qb[wdim + wave * 16 + 8 * half + r];

  #pragma unroll
  for (int nt = 0; nt < 8; ++nt) {
    int n  = nt * 16 + lr;
    int hh = h0 + (n >> 5), ww = n & 31;
    #pragma unroll
    for (int r = 0; r < 8; ++r) {
      int o = wave * 16 + 8 * half + r;
      out[(((b * DIM + o) << 5) + hh) * 32 + ww] = acc[nt][r] + bias[r];
    }
  }
}

extern "C" void kernel_launch(void* const* d_in, const int* in_sizes, int n_in,
                              void* d_out, int out_size, void* d_ws, size_t ws_size,
                              hipStream_t stream) {
  const float* q   = (const float*)d_in[0];   // question_rep (B, 128*128*9 + 128)
  const float* lhs = (const float*)d_in[1];   // (B,128,32,32)
  const float* rhs = (const float*)d_in[2];   // (B,128,32,32)
  const float* pw  = (const float*)d_in[3];   // (128,256)
  const float* pb  = (const float*)d_in[4];   // (128,)
  float* out = (float*)d_out;

  const int B = in_sizes[1] / (DIM * 32 * 32);
  char* ws = (char*)d_ws;
  _Float16* inp = (_Float16*)ws;                                  // B*1024*128 f16
  _Float16* wq  = (_Float16*)(ws + (size_t)B * 1024 * DIM * 2);   // B*9*128*128 f16
  _Float16* pwh = (_Float16*)(ws + (size_t)B * (1024 + 9 * DIM) * DIM * 2); // 128*256 f16

  wrepack_kernel<<<B * 576, 256, 0, stream>>>(q, wq);
  pwrepack_kernel<<<128, 256, 0, stream>>>(pw, pwh);
  proj_kernel<<<B * 8, 256, 0, stream>>>(lhs, rhs, pwh, pb, inp);
  conv_kernel<<<B * 8, 256, 0, stream>>>(q, wq, inp, out);
  (void)n_in; (void)out_size; (void)ws_size;
}
